// SelfAttention_377957122287
// MI455X (gfx1250) — matrix operand order repaired and checked by reading, not compiled, not run on hardware
//
#include <hip/hip_runtime.h>

typedef _Float16 f16;
typedef __attribute__((ext_vector_type(16))) _Float16 v16h;
typedef __attribute__((ext_vector_type(8)))  _Float16 v8h;
typedef __attribute__((ext_vector_type(4)))  _Float16 v4h;
typedef __attribute__((ext_vector_type(8)))  float    v8f;
typedef __attribute__((ext_vector_type(4)))  float    v4f;
typedef __attribute__((ext_vector_type(4)))  unsigned v4u;
typedef __attribute__((ext_vector_type(8)))  int      v8i;
typedef __attribute__((ext_vector_type(4)))  int      v4i;

union HFrag { v16h h; v8h h8[2]; };

// Problem constants
constexpr int Bn   = 2;
constexpr int T    = 4096;
constexpr int Cc   = 1024;
constexpr int Hh   = 16;
constexpr int Dh   = 64;
constexpr int Mtot = Bn * T;   // 8192
constexpr int Kdim = Cc;       // 1024 (GEMM K for both GEMMs)

// GEMM tiling
constexpr int BM  = 128;
constexpr int BN  = 128;
constexpr int BK  = 32;
constexpr int LDT = BK + 8;    // LDS tile row stride in halves (80B, 16B-aligned)

// ---------------- WMMA helpers (ISA 7.12.2 layouts) -----------------------
__device__ inline v8f wmma_f16(v16h a, v16h b, v8f c) {
  return __builtin_amdgcn_wmma_f32_16x16x32_f16(false, a, false, b,
                                                (short)0, c, false, false);
}

// A frag (16x32): lane&15 = M row; 16 contiguous k at 16*half
__device__ inline v16h load_a_frag(const f16* tile, int row, int lm, int half) {
  const f16* p = tile + (row + lm) * LDT;
  HFrag f;
  f.h8[0] = *(const v8h*)(p + 16 * half);
  f.h8[1] = *(const v8h*)(p + 16 * half + 8);
  return f.h;
}

// B frag (32x16) from [n][k] tile: lane&15 = N col; 16 contiguous k at 16*half
__device__ inline v16h load_b_frag(const f16* tile, int col, int lm, int half) {
  const f16* p = tile + (col + lm) * LDT + 16 * half;
  HFrag f;
  f.h8[0] = *(const v8h*)p;
  f.h8[1] = *(const v8h*)(p + 8);
  return f.h;
}

// ---------------- TDM tile load (CDNA5 Tensor Data Mover) -----------------
// Loads a 128x32-halves 2D tile (row stride = strideElems) into LDS with
// 4-DWORD padding every 16 DWORDs (matches LDT = 32+8 halves).
__device__ inline void tdm_load_tile(const f16* src, unsigned ldsOff,
                                     int rowsTotal, int strideElems) {
  unsigned long long ga = (unsigned long long)(uintptr_t)src;
  v4u g0;
  g0[0] = 1u;                                    // count=1 (valid user D#)
  g0[1] = ldsOff;                                // lds_addr (bytes)
  g0[2] = (unsigned)(ga & 0xFFFFFFFFu);          // global_addr[31:0]
  g0[3] = (unsigned)((ga >> 32) & 0x1FFFFFFu)    // global_addr[56:32]
          | 0x80000000u;                         // type=2 (bits 127:126)
  v8i g1;
  g1[0] = (1 << 16)            // data_size = 2 bytes
        | (1 << 20)            // pad_enable
        | (3 << 22)            // pad_interval: every 16 DWORDs
        | (3 << 25);           // pad_amount: 4 DWORDs
  g1[1] = (int)((unsigned)Kdim << 16);        // tensor_dim0[15:0] = 1024
  g1[2] = (int)(((unsigned)rowsTotal & 0xFFFFu) << 16);  // tensor_dim1 lo16
  g1[3] = (int)(32u << 16);                   // tile_dim0 = 32
  g1[4] = 128;                                // tile_dim1 = 128, tile_dim2 = 0
  g1[5] = strideElems;                        // tensor_dim0_stride lo32
  g1[6] = 0;
  g1[7] = 0;
  v4i z4 = {0, 0, 0, 0};
  v8i z8 = {0, 0, 0, 0, 0, 0, 0, 0};
  __builtin_amdgcn_tensor_load_to_lds(g0, g1, z4, z4, z8, 0);
}

__device__ inline unsigned lds_off_u32(const void* p) {
  return (unsigned)(uintptr_t)p;  // low 32 bits of generic LDS addr = offset
}

// ---------------- permlane16 xor-butterfly reductions (16-lane rows) ------
__device__ inline float xor16(float x, unsigned s0, unsigned s1) {
  return __uint_as_float(__builtin_amdgcn_permlane16(
      __float_as_uint(x), __float_as_uint(x), s0, s1, false, false));
}
__device__ inline float rowmax16(float x) {
  x = fmaxf(x, xor16(x, 0x67452301u, 0xEFCDAB89u));  // xor 1
  x = fmaxf(x, xor16(x, 0x54761032u, 0xDCFE98BAu));  // xor 2
  x = fmaxf(x, xor16(x, 0x32107654u, 0xBA98FEDCu));  // xor 4
  x = fmaxf(x, xor16(x, 0xFEDCBA98u, 0x76543210u));  // xor 8
  return x;
}
__device__ inline float rowsum16(float x) {
  x += xor16(x, 0x67452301u, 0xEFCDAB89u);
  x += xor16(x, 0x54761032u, 0xDCFE98BAu);
  x += xor16(x, 0x32107654u, 0xBA98FEDCu);
  x += xor16(x, 0xFEDCBA98u, 0x76543210u);
  return x;
}

// ---------------- one-time conversion kernels -----------------------------
__global__ void __launch_bounds__(256)
convert_f32_f16(const float* __restrict__ src, f16* __restrict__ dst) {
  int i = blockIdx.x * 256 + threadIdx.x;
  const v4f* s = (const v4f*)src + (size_t)i * 2;
  v4f a = s[0], b = s[1];
  v8h o;
  o[0] = (f16)a.x; o[1] = (f16)a.y; o[2] = (f16)a.z; o[3] = (f16)a.w;
  o[4] = (f16)b.x; o[5] = (f16)b.y; o[6] = (f16)b.z; o[7] = (f16)b.w;
  *(v8h*)(dst + (size_t)i * 8) = o;
}

// W[K,N] f32 -> Wt[N,K] f16 (32x32 tiles through LDS)
__global__ void __launch_bounds__(256)
transpose_f32_f16(const float* __restrict__ W, f16* __restrict__ Wt,
                  int K, int N) {
  __shared__ f16 ts[32][36];
  const int kb = (blockIdx.x % (K / 32)) * 32;
  const int nb = (blockIdx.x / (K / 32)) * 32;
  const int tid = threadIdx.x;
  const int r = tid >> 3, c4 = (tid & 7) * 4;
  v4f fv = *(const v4f*)(W + (size_t)(kb + r) * N + nb + c4);
  ts[r][c4 + 0] = (f16)fv.x; ts[r][c4 + 1] = (f16)fv.y;
  ts[r][c4 + 2] = (f16)fv.z; ts[r][c4 + 3] = (f16)fv.w;
  __syncthreads();
  v4h o;
  o[0] = ts[c4 + 0][r]; o[1] = ts[c4 + 1][r];
  o[2] = ts[c4 + 2][r]; o[3] = ts[c4 + 3][r];
  *(v4h*)(Wt + (size_t)(nb + r) * K + kb + c4) = o;
}

// ---------------- TDM double-buffered WMMA GEMM ---------------------------
// out = A[M,K](f16) @ Wt[N,K]^T(f16) + bias
// MODE 1: scatter f16 into q[B,H,T,D], k[B,H,T,D], v[B,H,D,T]
// MODE 0: f32 out[M,N]
template <int MODE>
__global__ void __launch_bounds__(256)
gemm_tdm(const f16* __restrict__ A, const f16* __restrict__ Bt,
         const float* __restrict__ bias, float* __restrict__ out32,
         f16* __restrict__ qb, f16* __restrict__ kb, f16* __restrict__ vb,
         int Ntot, int nblk) {
  __shared__ f16 As[2][BM * LDT];
  __shared__ f16 Bs[2][BN * LDT];

  const int tid  = threadIdx.x;
  const int m0   = (blockIdx.x / nblk) * BM;
  const int n0   = (blockIdx.x % nblk) * BN;
  const int lane = tid & 31, wave = tid >> 5;
  const int wm   = wave >> 1, wn = wave & 1;
  const int lm   = lane & 15, half = lane >> 4;

  v8f acc[2][4] = {};

  if (wave == 0) {
    tdm_load_tile(A + (size_t)m0 * Kdim, lds_off_u32(&As[0][0]), Mtot, Kdim);
    tdm_load_tile(Bt + (size_t)n0 * Kdim, lds_off_u32(&Bs[0][0]), Ntot, Kdim);
  }

  int buf = 0;
  for (int k0 = 0; k0 < Kdim; k0 += BK) {
    __syncthreads();  // prior compute on buf^1 finished -> safe to overwrite
    if (wave == 0) {
      if (k0 + BK < Kdim) {
        tdm_load_tile(A + (size_t)m0 * Kdim + k0 + BK,
                      lds_off_u32(&As[buf ^ 1][0]), Mtot, Kdim);
        tdm_load_tile(Bt + (size_t)n0 * Kdim + k0 + BK,
                      lds_off_u32(&Bs[buf ^ 1][0]), Ntot, Kdim);
        __builtin_amdgcn_s_wait_tensorcnt(2);  // current pair complete
      } else {
        __builtin_amdgcn_s_wait_tensorcnt(0);
      }
    }
    __syncthreads();  // tile ready for all waves

    v16h a0 = load_a_frag(As[buf], wm * 32, lm, half);
    v16h a1 = load_a_frag(As[buf], wm * 32 + 16, lm, half);
#pragma unroll
    for (int ni = 0; ni < 4; ++ni) {
      v16h b = load_b_frag(Bs[buf], wn * 64 + ni * 16, lm, half);
      acc[0][ni] = wmma_f16(a0, b, acc[0][ni]);
      acc[1][ni] = wmma_f16(a1, b, acc[1][ni]);
    }
    buf ^= 1;
  }

#pragma unroll
  for (int mi = 0; mi < 2; ++mi) {
#pragma unroll
    for (int ni = 0; ni < 4; ++ni) {
#pragma unroll
      for (int r = 0; r < 8; ++r) {
        int m = m0 + wm * 32 + mi * 16 + r + 8 * half;
        int n = n0 + wn * 64 + ni * 16 + lm;
        float val = acc[mi][ni][r] + bias[n];
        if constexpr (MODE == 0) {
          out32[(size_t)m * Ntot + n] = val;
        } else {
          int which = n >> 10;
          int cc = n & 1023;
          int hh = cc >> 6, dd = cc & 63;
          int bb = m >> 12, tt = m & 4095;
          int bh = bb * Hh + hh;
          f16 hv = (f16)val;
          if (which == 0)      qb[((size_t)bh * T + tt) * Dh + dd] = hv;
          else if (which == 1) kb[((size_t)bh * T + tt) * Dh + dd] = hv;
          else                 vb[((size_t)bh * Dh + dd) * T + tt] = hv;
        }
      }
    }
  }
}

// ---------------- Flash attention (one wave = 16 query rows) --------------
__global__ void __launch_bounds__(128)
attn_kernel(const f16* __restrict__ q, const f16* __restrict__ kkv,
            const f16* __restrict__ vvv, f16* __restrict__ y) {
  __shared__ f16 Ps[4 * 16 * 40];  // per-wave 16x32 P scratch, stride 40
  const int tid  = threadIdx.x;
  const int wave = tid >> 5, lane = tid & 31;
  const int lm   = lane & 15, half = lane >> 4;
  const int wg   = blockIdx.x * 4 + wave;
  const int bh   = wg >> 8;
  const int t0   = (wg & 255) << 4;
  f16* P = Ps + wave * (16 * 40);

  // q tile 16x64 -> two A frags
  v16h aq[2];
  const f16* qrow = q + ((size_t)bh * T + t0 + lm) * Dh;
#pragma unroll
  for (int ks = 0; ks < 2; ++ks) {
    HFrag f;
    f.h8[0] = *(const v8h*)(qrow + 32 * ks + 16 * half);
    f.h8[1] = *(const v8h*)(qrow + 32 * ks + 16 * half + 8);
    aq[ks] = f.h;
  }

  float rmax[8], rsum[8];
  v8f o[4] = {};
#pragma unroll
  for (int r = 0; r < 8; ++r) { rmax[r] = -1e30f; rsum[r] = 0.f; }

  const f16* kbase = kkv + (size_t)bh * T * Dh;
  const f16* vbase = vvv + (size_t)bh * Dh * T;

  for (int s0 = 0; s0 < T; s0 += 32) {
    if (s0 + 32 < T) {
      __builtin_prefetch(kbase + (size_t)(s0 + 32 + lm) * Dh, 0, 1);
      __builtin_prefetch(vbase + (size_t)lm * T + s0 + 32, 0, 1);
    }
    // S = q @ k^T (16x32 block)
    v8f sacc[2];
#pragma unroll
    for (int nt = 0; nt < 2; ++nt) {
      v8f sa = {};
#pragma unroll
      for (int ks = 0; ks < 2; ++ks) {
        const f16* kp = kbase + (size_t)(s0 + nt * 16 + lm) * Dh + 32 * ks +
                        16 * half;
        HFrag f;
        f.h8[0] = *(const v8h*)kp;
        f.h8[1] = *(const v8h*)(kp + 8);
        sa = wmma_f16(aq[ks], f.h, sa);
      }
      sacc[nt] = sa;
    }

    // online softmax per row (16-lane-half confined permlane reductions)
#pragma unroll
    for (int r = 0; r < 8; ++r) {
      float x0 = sacc[0][r] * 0.125f;
      float x1 = sacc[1][r] * 0.125f;
      float mnew = fmaxf(rmax[r], rowmax16(fmaxf(x0, x1)));
      float corr = __expf(rmax[r] - mnew);
      rmax[r] = mnew;
      float p0 = __expf(x0 - mnew);
      float p1 = __expf(x1 - mnew);
      rsum[r] = rsum[r] * corr + rowsum16(p0 + p1);
#pragma unroll
      for (int t = 0; t < 4; ++t) o[t][r] *= corr;
      int mrow = r + 8 * half;
      P[mrow * 40 + lm]      = (f16)p0;
      P[mrow * 40 + 16 + lm] = (f16)p1;
    }
    asm volatile("s_wait_dscnt 0" ::: "memory");

    // P (C-layout) -> A frag via per-wave LDS
    HFrag pf;
    pf.h8[0] = *(const v8h*)(P + lm * 40 + 16 * half);
    pf.h8[1] = *(const v8h*)(P + lm * 40 + 16 * half + 8);

    // o += P @ V (V is [d][t] so B frags are contiguous)
#pragma unroll
    for (int t = 0; t < 4; ++t) {
      const f16* vp = vbase + (size_t)(t * 16 + lm) * T + s0 + 16 * half;
      HFrag bf;
      bf.h8[0] = *(const v8h*)vp;
      bf.h8[1] = *(const v8h*)(vp + 8);
      o[t] = wmma_f16(pf.h, bf.h, o[t]);
    }
  }

  const int bb = bh >> 4, hh = bh & 15;
#pragma unroll
  for (int r = 0; r < 8; ++r) {
    float inv = 1.0f / rsum[r];
    int row = bb * T + t0 + r + 8 * half;
#pragma unroll
    for (int t = 0; t < 4; ++t)
      y[(size_t)row * Cc + hh * Dh + t * 16 + lm] = (f16)(o[t][r] * inv);
  }
}

extern "C" void kernel_launch(void* const* d_in, const int* in_sizes, int n_in,
                              void* d_out, int out_size, void* d_ws,
                              size_t ws_size, hipStream_t stream) {
  const float* x      = (const float*)d_in[0];
  const float* W_attn = (const float*)d_in[1];
  const float* b_attn = (const float*)d_in[2];
  const float* W_proj = (const float*)d_in[3];
  const float* b_proj = (const float*)d_in[4];
  float* out = (float*)d_out;

  const size_t qElems = (size_t)Bn * Hh * T * Dh;  // 8388608
  f16* qb  = (f16*)d_ws;
  f16* kb  = qb + qElems;
  f16* vb  = kb + qElems;
  f16* yb  = vb + qElems;
  f16* xh  = yb + qElems;
  f16* WaT = xh + qElems;                 // [3C, C]
  f16* WpT = WaT + (size_t)3 * Cc * Cc;   // [C, C]

  // one-time f16 conversion / weight transpose
  convert_f32_f16<<<dim3((Mtot * Cc) / (256 * 8)), dim3(256), 0, stream>>>(x, xh);
  transpose_f32_f16<<<dim3((Cc / 32) * (3 * Cc / 32)), dim3(256), 0, stream>>>(
      W_attn, WaT, Cc, 3 * Cc);
  transpose_f32_f16<<<dim3((Cc / 32) * (Cc / 32)), dim3(256), 0, stream>>>(
      W_proj, WpT, Cc, Cc);

  // QKV GEMM (TDM-fed): [8192,1024] @ [1024,3072] -> q/k/v f16
  gemm_tdm<1><<<dim3((3 * Cc / BN) * (Mtot / BM)), dim3(256), 0, stream>>>(
      xh, WaT, b_attn, nullptr, qb, kb, vb, 3 * Cc, 3 * Cc / BN);

  // Flash attention
  attn_kernel<<<dim3(Bn * Hh * (T / 16) / 4), dim3(128), 0, stream>>>(qb, kb,
                                                                      vb, yb);

  // Proj GEMM (TDM-fed): [8192,1024] @ [1024,1024] + b -> f32
  gemm_tdm<0><<<dim3((Cc / BN) * (Mtot / BM)), dim3(256), 0, stream>>>(
      yb, WpT, b_proj, out, nullptr, nullptr, nullptr, Cc, Cc / BN);
}
